// ShiftedWindowAttentionV2_76570676953310
// MI455X (gfx1250) — compile-verified, matrix operand order checked by reading, not run on hardware
//
#include <hip/hip_runtime.h>
#include <hip/hip_bf16.h>
#include <math.h>

// ---------------------------------------------------------------------------
// SwinV2 shifted-window attention, fused per-window, CDNA5 (gfx1250) WMMA bf16
// v2.1: double-buffered weight staging (1 barrier/k-step), parallel softmax.
//       (LDS double-buffer halves selected arithmetically -- no pointer array,
//        which previously produced an unlinkable addrspacecast initializer.)
// ---------------------------------------------------------------------------
typedef __bf16 bf16;
typedef bf16  v16bf __attribute__((ext_vector_type(16)));
typedef bf16  v8bf  __attribute__((ext_vector_type(8)));
typedef float v8f   __attribute__((ext_vector_type(8)));

#define WMMA_BF16(a, b, c) \
  __builtin_amdgcn_wmma_f32_16x16x32_bf16(false, (a), false, (b), (short)0, (c), false, false)

// Load a 16-element bf16 WMMA A/B fragment from a row-major [rows x stride]
// LDS matrix. Caller passes p = base + (tile_row0 + lane%16)*stride + k0 + kk
// where kk = (lane<16 ? 0 : 8). Elements 0..7 <- K kk..kk+7, 8..15 <- K kk+16..kk+23
// (ISA 7.12.2 16-bit A 16x32 layout; B uses the same pattern with row = column).
__device__ __forceinline__ v16bf ldfrag(const bf16* p) {
  v8bf lo = *(const v8bf*)(p);
  v8bf hi = *(const v8bf*)(p + 16);
  v16bf f;
#pragma unroll
  for (int i = 0; i < 8; ++i) { f[i] = lo[i]; f[8 + i] = hi[i]; }
  return f;
}

__device__ __forceinline__ int region1(int c) { return (c < 56) ? 0 : ((c < 60) ? 1 : 2); }

// ---------------------------------------------------------------------------
// Kernel 1: continuous-position-bias MLP -> 16*sigmoid table (12, 64, 64) in ws
// ---------------------------------------------------------------------------
__global__ __launch_bounds__(256) void swin_cpb_bias_kernel(
    const float* __restrict__ w1, const float* __restrict__ b1,
    const float* __restrict__ w2, float* __restrict__ biasws) {
  __shared__ float tab[225 * 12];
  const int tid = threadIdx.x;
  for (int p = tid; p < 225; p += 256) {
    int i = p / 15, j = p % 15;
    float t0 = (float)(i - 7) * (8.0f / 7.0f);
    float t1 = (float)(j - 7) * (8.0f / 7.0f);
    t0 = ((t0 > 0.f) ? 1.f : ((t0 < 0.f) ? -1.f : 0.f)) * log2f(fabsf(t0) + 1.f) * (1.f / 3.f);
    t1 = ((t1 > 0.f) ? 1.f : ((t1 < 0.f) ? -1.f : 0.f)) * log2f(fabsf(t1) + 1.f) * (1.f / 3.f);
    float acc[12];
#pragma unroll
    for (int hh = 0; hh < 12; ++hh) acc[hh] = 0.f;
    for (int k = 0; k < 512; ++k) {
      float hk = fmaxf(t0 * w1[2 * k] + t1 * w1[2 * k + 1] + b1[k], 0.f);
#pragma unroll
      for (int hh = 0; hh < 12; ++hh) acc[hh] += hk * w2[hh * 512 + k];
    }
#pragma unroll
    for (int hh = 0; hh < 12; ++hh) tab[p * 12 + hh] = acc[hh];
  }
  __syncthreads();
  for (int e = tid; e < 12 * 64 * 64; e += 256) {
    int hh = e >> 12;
    int r = e & 4095;
    int i = r >> 6, j = r & 63;
    int d0 = (i >> 3) - (j >> 3) + 7;
    int d1 = (i & 7) - (j & 7) + 7;
    float v = tab[(d0 * 15 + d1) * 12 + hh];
    biasws[e] = 16.f / (1.f + __expf(-v));  // 16 * sigmoid
  }
}

// ---------------------------------------------------------------------------
// Kernel 2: one block per window (1024 blocks, 8 wave32 waves).
// ---------------------------------------------------------------------------
__global__ __launch_bounds__(256) void swin_attn_kernel(
    const float* __restrict__ x, const float* __restrict__ qkv_w,
    const float* __restrict__ qkv_b, const float* __restrict__ proj_w,
    const float* __restrict__ proj_b, const float* __restrict__ logit_scale,
    const float* __restrict__ biasws, float* __restrict__ out) {

  __shared__ bf16  xs[64 * 384];      // 48 KB: rolled window; later 2x24KB proj W buffers
  __shared__ bf16  outb[64 * 384];    // 48 KB: per-window attention output (A of proj)
  __shared__ bf16  wtile[2][96 * 32]; // 12 KB: double-buffered QKV weight tiles
  __shared__ bf16  qb[64 * 32];       //  4 KB: normalized Q head tile
  __shared__ bf16  kb[64 * 32];       //  4 KB: normalized K head tile
  __shared__ bf16  vtr[32 * 64];      //  4 KB: V head tile, transposed [dim][token]
  __shared__ float Sbuf[64 * 64];     // 16 KB: attention logits (fp32 for softmax)
  __shared__ bf16  Pbuf[64 * 64];     //  8 KB: softmax probabilities (A of P@V)
  __shared__ float redmax[256];       //  1 KB: softmax partial maxima
  __shared__ float redsum[256];       //  1 KB: softmax partial sums

  const int tid = threadIdx.x;
  const int wave = tid >> 5;
  const int lane = tid & 31;
  const int lr = lane & 15;
  const int kk = (lane < 16) ? 0 : 8;       // K split for A/B fragments
  const int hi_half = (lane < 16) ? 0 : 8;  // M offset for C/D fragments

  const int blk = blockIdx.x;
  const int bb = blk >> 6;   // batch
  const int win = blk & 63;  // window in 8x8 grid
  const int wy = win >> 3, wx = win & 7;

  // -------- Phase 0: load rolled window (roll -4,-4 + partition) as bf16 ------
  for (int chunk = tid; chunk < 64 * 96; chunk += 256) {
    int t = chunk / 96;   // token (iy*8+ix)
    int c4 = chunk % 96;  // float4 chunk along C
    int sh = ((wy << 3) + (t >> 3) + 4) & 63;
    int sw = ((wx << 3) + (t & 7) + 4) & 63;
    const float4 v = *(const float4*)(x + (((size_t)bb * 64 + sh) * 64 + sw) * 384 + c4 * 4);
    bf16* dst = xs + t * 384 + c4 * 4;
    dst[0] = (bf16)v.x; dst[1] = (bf16)v.y; dst[2] = (bf16)v.z; dst[3] = (bf16)v.w;
  }
  __syncthreads();

  // ---------------------------- per-head pipeline ----------------------------
  for (int h = 0; h < 12; ++h) {
    // QKV weight tile staging: o 0..31 -> q, 32..63 -> k, 64..95 -> v rows of qkv_w
    auto stage_qkv = [&](int ks, bf16* dst) {
      for (int idx = tid; idx < 96 * 32; idx += 256) {
        int o = idx >> 5, kc = idx & 31;
        int g = (o < 32) ? (h * 32 + o)
                         : ((o < 64) ? (384 + h * 32 + (o - 32)) : (768 + h * 32 + (o - 64)));
        dst[idx] = (bf16)qkv_w[g * 384 + ks * 32 + kc];
        if (ks + 1 < 12) __builtin_prefetch(qkv_w + g * 384 + (ks + 1) * 32 + kc, 0, 0);
      }
    };

    // ---- QKV head GEMM: [64 x 96] = xs[64x384] @ Wh^T, 12 k-steps of 32,
    //      double-buffered: one barrier per k-step, staging overlaps WMMA ----
    v8f acc[3];
#pragma unroll
    for (int i = 0; i < 3; ++i)
#pragma unroll
      for (int v = 0; v < 8; ++v) acc[i][v] = 0.f;

    const int tt0 = wave * 3;  // 24 tiles (4 Mt x 6 Nt) over 8 waves
    stage_qkv(0, wtile[0]);
    for (int ks = 0; ks < 12; ++ks) {
      __syncthreads();  // publishes buffer ks&1; protects buffer (ks+1)&1 (read in ks-1)
      const bf16* cur = wtile[ks & 1];
      if (ks + 1 < 12) stage_qkv(ks + 1, wtile[(ks + 1) & 1]);
#pragma unroll
      for (int i = 0; i < 3; ++i) {
        int tt = tt0 + i, mt = tt / 6, nt = tt % 6;
        v16bf a  = ldfrag(xs + (mt * 16 + lr) * 384 + ks * 32 + kk);
        v16bf bm = ldfrag(cur + (nt * 16 + lr) * 32 + kk);
        acc[i] = WMMA_BF16(a, bm, acc[i]);
      }
    }
    __syncthreads();

    // ---- add qkv bias (k-bias zeroed, SwinV2) and scatter to qb/kb/vtr ----
#pragma unroll
    for (int i = 0; i < 3; ++i) {
      int tt = tt0 + i, mt = tt / 6, nt = tt % 6;
      int n = nt * 16 + lr;  // 0..95
      float bia = (n < 32) ? qkv_b[h * 32 + n]
                           : ((n < 64) ? 0.f : qkv_b[768 + h * 32 + (n - 64)]);
#pragma unroll
      for (int v = 0; v < 8; ++v) {
        int m = mt * 16 + v + hi_half;
        float val = acc[i][v] + bia;
        if (n < 32)      qb[m * 32 + n] = (bf16)val;
        else if (n < 64) kb[m * 32 + (n - 32)] = (bf16)val;
        else             vtr[(n - 64) * 64 + m] = (bf16)val;  // transposed store
      }
    }
    __syncthreads();

    // ---- cosine-attn row norms; fold exp(min(logit_scale, ln100)) into Q ----
    if (tid < 128) {
      int which = tid >> 6, r = tid & 63;
      bf16* row = which ? (kb + r * 32) : (qb + r * 32);
      float s = 0.f;
      for (int j = 0; j < 32; ++j) { float xv = (float)row[j]; s += xv * xv; }
      float inv = 1.f / fmaxf(sqrtf(s), 1e-12f);
      if (!which) inv *= __expf(fminf(logit_scale[h], 4.605170185988091f));  // ln(100)
      for (int j = 0; j < 32; ++j) row[j] = (bf16)((float)row[j] * inv);
    }
    __syncthreads();

    // ---- S = Qn @ Kn^T (K=32) + CPB bias + analytic shift mask ----
#pragma unroll
    for (int i = 0; i < 2; ++i) {  // 16 tiles (4x4) over 8 waves
      int tt = wave * 2 + i, mt = tt >> 2, nt = tt & 3;
      v8f s;
#pragma unroll
      for (int v = 0; v < 8; ++v) s[v] = 0.f;
      v16bf a  = ldfrag(qb + (mt * 16 + lr) * 32 + kk);
      v16bf bm = ldfrag(kb + (nt * 16 + lr) * 32 + kk);  // B = Kn^T gather
      s = WMMA_BF16(a, bm, s);
      int n = nt * 16 + lr;
      int cn = region1((wy << 3) + (n >> 3)) * 3 + region1((wx << 3) + (n & 7));
#pragma unroll
      for (int v = 0; v < 8; ++v) {
        int m = mt * 16 + v + hi_half;
        int cm = region1((wy << 3) + (m >> 3)) * 3 + region1((wx << 3) + (m & 7));
        float val = s[v] + biasws[(h * 64 + m) * 64 + n] + ((cm != cn) ? -100.f : 0.f);
        Sbuf[m * 64 + n] = val;
      }
    }
    __syncthreads();

    // ---- row softmax, 4 threads per row (all 256 threads busy) ----
    {
      int r = tid >> 2, qd = tid & 3;
      const float* seg = Sbuf + r * 64 + qd * 16;
      float mx = -1e30f;
#pragma unroll
      for (int j = 0; j < 16; ++j) mx = fmaxf(mx, seg[j]);
      redmax[tid] = mx;
      __syncthreads();
      float m4 = fmaxf(fmaxf(redmax[r * 4 + 0], redmax[r * 4 + 1]),
                       fmaxf(redmax[r * 4 + 2], redmax[r * 4 + 3]));
      float sum = 0.f;
#pragma unroll
      for (int j = 0; j < 16; ++j) sum += __expf(seg[j] - m4);
      redsum[tid] = sum;
      __syncthreads();
      float rinv = 1.f / (redsum[r * 4 + 0] + redsum[r * 4 + 1] +
                          redsum[r * 4 + 2] + redsum[r * 4 + 3]);
#pragma unroll
      for (int j = 0; j < 16; ++j)
        Pbuf[r * 64 + qd * 16 + j] = (bf16)(__expf(seg[j] - m4) * rinv);
    }
    __syncthreads();

    // ---- O_h = P @ V : [64 x 32], K=64 (2 k-steps), 8 tiles over 8 waves ----
    {
      int mt = wave >> 1, nt = wave & 1;
      v8f o;
#pragma unroll
      for (int v = 0; v < 8; ++v) o[v] = 0.f;
#pragma unroll
      for (int k2 = 0; k2 < 64; k2 += 32) {
        v16bf a  = ldfrag(Pbuf + (mt * 16 + lr) * 64 + k2 + kk);
        v16bf bm = ldfrag(vtr + (nt * 16 + lr) * 64 + k2 + kk);  // B[k][n] = vtr[n][k]
        o = WMMA_BF16(a, bm, o);
      }
      int n = nt * 16 + lr;
#pragma unroll
      for (int v = 0; v < 8; ++v) {
        int m = mt * 16 + v + hi_half;
        outb[m * 384 + h * 32 + n] = (bf16)o[v];
      }
    }
    __syncthreads();  // protect wtile/qb/kb/vtr/Pbuf before next head restages
  }  // heads

  // ---- Projection: out = outb[64x384] @ proj_w^T + proj_b; 96 tiles ----
  // xs is dead here: reuse it as a 2 x 24 KB double buffer for proj_w tiles.
  // NOTE: select halves arithmetically -- an LDS pointer array initializer
  // generates an addrspacecast constant ld.lld cannot link.
  auto projbuf = [&](int which) -> bf16* { return xs + which * (384 * 32); };
  auto stage_proj = [&](int ks, bf16* dst) {
    for (int idx = tid; idx < 384 * 32; idx += 256) {
      int o = idx >> 5, kc = idx & 31;
      dst[idx] = (bf16)proj_w[o * 384 + ks * 32 + kc];
      if (ks + 1 < 12) __builtin_prefetch(proj_w + o * 384 + (ks + 1) * 32 + kc, 0, 0);
    }
  };

  v8f pacc[12];
#pragma unroll
  for (int i = 0; i < 12; ++i)
#pragma unroll
    for (int v = 0; v < 8; ++v) pacc[i][v] = 0.f;
  const int pmt = wave >> 1;         // M tile (each covered by 2 waves)
  const int pnt0 = (wave & 1) * 12;  // N tile base
  stage_proj(0, projbuf(0));
  for (int ks = 0; ks < 12; ++ks) {
    __syncthreads();
    const bf16* cur = projbuf(ks & 1);
    if (ks + 1 < 12) stage_proj(ks + 1, projbuf((ks + 1) & 1));
    v16bf a = ldfrag(outb + (pmt * 16 + lr) * 384 + ks * 32 + kk);
#pragma unroll
    for (int i = 0; i < 12; ++i) {
      v16bf bm = ldfrag(cur + ((pnt0 + i) * 16 + lr) * 32 + kk);
      pacc[i] = WMMA_BF16(a, bm, pacc[i]);
    }
  }

  // ---- epilogue: + proj_b, window-reverse + roll(+4,+4), store fp32 ----
#pragma unroll
  for (int i = 0; i < 12; ++i) {
    int c = (pnt0 + i) * 16 + lr;
    float pb = proj_b[c];
#pragma unroll
    for (int v = 0; v < 8; ++v) {
      int m = pmt * 16 + v + hi_half;
      int dh = ((wy << 3) + (m >> 3) + 4) & 63;
      int dw = ((wx << 3) + (m & 7) + 4) & 63;
      out[(((size_t)bb * 64 + dh) * 64 + dw) * 384 + c] = pacc[i][v] + pb;
    }
  }
}

// ---------------------------------------------------------------------------
extern "C" void kernel_launch(void* const* d_in, const int* in_sizes, int n_in,
                              void* d_out, int out_size, void* d_ws, size_t ws_size,
                              hipStream_t stream) {
  const float* x      = (const float*)d_in[0];
  const float* qkv_w  = (const float*)d_in[1];
  const float* qkv_b  = (const float*)d_in[2];
  const float* proj_w = (const float*)d_in[3];
  const float* proj_b = (const float*)d_in[4];
  const float* lsc    = (const float*)d_in[5];
  const float* cpb_w1 = (const float*)d_in[6];
  const float* cpb_b1 = (const float*)d_in[7];
  const float* cpb_w2 = (const float*)d_in[8];
  float* out = (float*)d_out;
  float* biasws = (float*)d_ws;  // 12*64*64 floats = 192 KB

  swin_cpb_bias_kernel<<<dim3(1), dim3(256), 0, stream>>>(cpb_w1, cpb_b1, cpb_w2, biasws);
  swin_attn_kernel<<<dim3(1024), dim3(256), 0, stream>>>(x, qkv_w, qkv_b, proj_w, proj_b,
                                                         lsc, biasws, out);
}
